// Deform_Block_48232482734880
// MI455X (gfx1250) — matrix operand order repaired and checked by reading, not compiled, not run on hardware
//
#include <hip/hip_runtime.h>
#include <math.h>

// ---------------------------------------------------------------------------
// Types for CDNA5 WMMA
// ---------------------------------------------------------------------------
typedef __attribute__((ext_vector_type(16))) __bf16 v16bf;
typedef __attribute__((ext_vector_type(8)))  float  v8f;

#define EPSV 1e-5f

__device__ __forceinline__ unsigned short f2bf(float f) {
    union { float f; unsigned u; } v; v.f = f;
    unsigned u = v.u;
    u += 0x7fffu + ((u >> 16) & 1u);           // round-to-nearest-even
    return (unsigned short)(u >> 16);
}

__device__ __forceinline__ int clampi(int x, int lo, int hi) {
    return x < lo ? lo : (x > hi ? hi : x);
}

// ---------------------------------------------------------------------------
// 1) offset + mask 3x3 convs (27 output channels), pad=1
//    OM[pix][0..8]=off_x, [9..17]=off_y, [18..26]=sigmoid(mask)
// ---------------------------------------------------------------------------
__global__ void offmask_kernel(const float* __restrict__ I,
                               const float* __restrict__ wp, const float* __restrict__ bp,
                               const float* __restrict__ wm, const float* __restrict__ bm,
                               float* __restrict__ OM,
                               int B, int C, int H, int W, int h, int w, int stride)
{
    int M = B * h * w;
    int idx = blockIdx.x * blockDim.x + threadIdx.x;
    if (idx >= M * 27) return;
    int pix = idx / 27, oc = idx % 27;
    int b  = pix / (h * w);
    int ij = pix % (h * w);
    int i = ij / w, j = ij % w;
    const float* wgt = (oc < 18) ? (wp + (size_t)oc * C * 9)
                                 : (wm + (size_t)(oc - 18) * C * 9);
    int i0 = i * stride - 1, j0 = j * stride - 1;
    float sum = 0.f;
    for (int c = 0; c < C; ++c) {
        const float* Ic = I + ((size_t)b * C + c) * H * W;
        const float* wc = wgt + c * 9;
        #pragma unroll
        for (int di = 0; di < 3; ++di) {
            int ii = i0 + di;
            if (ii < 0 || ii >= H) continue;
            #pragma unroll
            for (int dj = 0; dj < 3; ++dj) {
                int jj = j0 + dj;
                if (jj < 0 || jj >= W) continue;
                sum += Ic[(size_t)ii * W + jj] * wc[di * 3 + dj];
            }
        }
    }
    float val;
    if (oc < 18) {
        val = sum + bp[oc];
    } else {
        float s = sum + bm[oc - 18];
        val = 1.f / (1.f + expf(-s));
    }
    OM[(size_t)pix * 27 + oc] = val;
}

// ---------------------------------------------------------------------------
// 2) bilinear gather + modulation, writing bf16 A-matrix in WMMA lane layout.
//    A element (pix, k=c*9+n) -> chunk = k/32, kc = k%32:
//      lane = (pix%16) + 16*((kc>>3)&1)
//      slot = (kc&7) | (((kc>>4)&1)<<3)
//    Asw[((Mt*nK32 + chunk)*32 + lane)*16 + slot]
// ---------------------------------------------------------------------------
__global__ void gather_kernel(const float* __restrict__ I, const float* __restrict__ OM,
                              unsigned short* __restrict__ Asw,
                              int B, int C, int H, int W, int h, int w, int stride, int nK32)
{
    int M = B * h * w;
    int idx = blockIdx.x * blockDim.x + threadIdx.x;
    if (idx >= M * 9) return;
    int pix = idx / 9, n = idx % 9;
    int di = n / 3, dj = n % 3;
    int b  = pix / (h * w);
    int ij = pix % (h * w);
    int i = ij / w, j = ij % w;

    float offx = OM[(size_t)pix * 27 + n];
    float offy = OM[(size_t)pix * 27 + 9 + n];
    float mval = OM[(size_t)pix * 27 + 18 + n];

    int Hp = H + 2, Wp = W + 2;
    float px = offx + (float)(di - 1) + (float)(1 + i * stride);
    float py = offy + (float)(dj - 1) + (float)(1 + j * stride);
    float fx = floorf(px), fy = floorf(py);
    int ltx = clampi((int)fx, 0, Hp - 1);
    int lty = clampi((int)fy, 0, Wp - 1);
    int rbx = clampi((int)fx + 1, 0, Hp - 1);
    int rby = clampi((int)fy + 1, 0, Wp - 1);
    float pxc = fminf(fmaxf(px, 0.f), (float)(Hp - 1));
    float pyc = fminf(fmaxf(py, 0.f), (float)(Wp - 1));
    float glt = (1.f + ((float)ltx - pxc)) * (1.f + ((float)lty - pyc));
    float grb = (1.f - ((float)rbx - pxc)) * (1.f - ((float)rby - pyc));
    float glb = (1.f + ((float)ltx - pxc)) * (1.f - ((float)rby - pyc));
    float grt = (1.f - ((float)rbx - pxc)) * (1.f + ((float)lty - pyc));

    bool vlx = (ltx >= 1 && ltx <= H), vly = (lty >= 1 && lty <= W);
    bool vrx = (rbx >= 1 && rbx <= H), vry = (rby >= 1 && rby <= W);
    float w_lt = (vlx && vly) ? glt : 0.f;
    float w_rb = (vrx && vry) ? grb : 0.f;
    float w_lb = (vlx && vry) ? glb : 0.f;
    float w_rt = (vrx && vly) ? grt : 0.f;
    int o_lt = (vlx && vly) ? ((ltx - 1) * W + (lty - 1)) : 0;
    int o_rb = (vrx && vry) ? ((rbx - 1) * W + (rby - 1)) : 0;
    int o_lb = (vlx && vry) ? ((ltx - 1) * W + (rby - 1)) : 0;
    int o_rt = (vrx && vly) ? ((rbx - 1) * W + (lty - 1)) : 0;

    int    laneBase = pix & 15;
    size_t tileBase = (size_t)(pix >> 4) * nK32;

    for (int c = 0; c < C; ++c) {
        const float* Ic = I + ((size_t)b * C + c) * H * W;
        float v = w_lt * Ic[o_lt] + w_rb * Ic[o_rb]
                + w_lb * Ic[o_lb] + w_rt * Ic[o_rt];
        v *= mval;
        int k   = c * 9 + n;
        int kc  = k & 31, c32 = k >> 5;
        int lane = laneBase + (((kc >> 3) & 1) << 4);
        int slot = (kc & 7) | (((kc >> 4) & 1) << 3);
        Asw[((tileBase + c32) * 32 + lane) * 16 + slot] = f2bf(v);
    }
}

// ---------------------------------------------------------------------------
// 3) weight prep: Wc(O=128, C, 9) -> bf16 B-matrix in WMMA lane layout.
//    B element (k, o): chunk = k/32, kc = k%32:
//      lane = (o%16) + 16*((kc>>3)&1), slot = (kc&7) | (((kc>>4)&1)<<3), colT = o/16
//    Bsw[((chunk*8 + colT)*32 + lane)*16 + slot]
// ---------------------------------------------------------------------------
__global__ void wprep_kernel(const float* __restrict__ Wc, unsigned short* __restrict__ Bsw,
                             int K, int C)
{
    int idx = blockIdx.x * blockDim.x + threadIdx.x;
    if (idx >= K * 128) return;
    int k = idx >> 7, o = idx & 127;
    int c = k / 9, n = k % 9;
    float v = Wc[((size_t)o * C + c) * 9 + n];
    int kc = k & 31, c32 = k >> 5;
    int lane = (o & 15) + (((kc >> 3) & 1) << 4);
    int slot = (kc & 7) | (((kc >> 4) & 1) << 3);
    int colT = o >> 4;
    Bsw[(((size_t)c32 * 8 + colT) * 32 + lane) * 16 + slot] = f2bf(v);
}

// ---------------------------------------------------------------------------
// 4) GEMM: Out[M][128] = A[M][K] * B[K][128] via v_wmma_f32_16x16x32_bf16.
//    One wave per 16x16 tile; 8 waves/block cover N=128; grid = M/16 tiles.
// ---------------------------------------------------------------------------
__global__ void __launch_bounds__(256)
gemm_wmma_kernel(const unsigned short* __restrict__ Asw,
                 const unsigned short* __restrict__ Bsw,
                 float* __restrict__ Out, int nK32)
{
    int lane = threadIdx.x;   // 0..31, one wave32 per threadIdx.y row
    int colT = threadIdx.y;   // 0..7
    int Mt   = blockIdx.x;

    v8f acc = {};
    const unsigned short* aPtr = Asw + ((size_t)Mt * nK32 * 32 + lane) * 16;
    const unsigned short* bPtr = Bsw + (((size_t)colT) * 32 + lane) * 16;

    for (int kc = 0; kc < nK32; ++kc) {
        v16bf a = *(const v16bf*)aPtr;
        v16bf b = *(const v16bf*)bPtr;
        acc = __builtin_amdgcn_wmma_f32_16x16x32_bf16(
            false, a, false, b, (short)0, acc, false, false);
        aPtr += 32 * 16;        // next 32-K chunk of this M-tile
        bPtr += 8 * 32 * 16;    // next 32-K chunk (8 col-tiles per chunk)
    }

    int row0 = Mt * 16 + ((lane >> 4) << 3);  // C/D layout: VGPR r -> M=r (+8 for hi lanes)
    int col  = colT * 16 + (lane & 15);
    #pragma unroll
    for (int r = 0; r < 8; ++r)
        Out[(size_t)(row0 + r) * 128 + col] = acc[r];
}

// ---------------------------------------------------------------------------
// 5) BN statistics: one block per channel, mean & population variance over M.
// ---------------------------------------------------------------------------
__global__ void stats_kernel(const float* __restrict__ Conv, float* __restrict__ stats, int M)
{
    int ch = blockIdx.x;
    float s = 0.f, sq = 0.f;
    for (int m = threadIdx.x; m < M; m += blockDim.x) {
        float v = Conv[(size_t)m * 128 + ch];
        s += v; sq += v * v;
    }
    __shared__ float sh_s[256], sh_q[256];
    sh_s[threadIdx.x] = s; sh_q[threadIdx.x] = sq;
    __syncthreads();
    for (int st = blockDim.x / 2; st > 0; st >>= 1) {
        if (threadIdx.x < (unsigned)st) {
            sh_s[threadIdx.x] += sh_s[threadIdx.x + st];
            sh_q[threadIdx.x] += sh_q[threadIdx.x + st];
        }
        __syncthreads();
    }
    if (threadIdx.x == 0) {
        float mean = sh_s[0] / (float)M;
        float var  = sh_q[0] / (float)M - mean * mean;
        stats[ch * 2]     = mean;
        stats[ch * 2 + 1] = var;
    }
}

// ---------------------------------------------------------------------------
// 6) BN + ReLU, write NCHW (input to next DCN stage)
// ---------------------------------------------------------------------------
__global__ void bnrelu_kernel(const float* __restrict__ Conv, const float* __restrict__ stats,
                              const float* __restrict__ gam, const float* __restrict__ bet,
                              float* __restrict__ Y, int B, int h, int w)
{
    int M = B * h * w;
    int idx = blockIdx.x * blockDim.x + threadIdx.x;
    if (idx >= M * 128) return;
    int pix = idx >> 7, o = idx & 127;
    float v = Conv[idx];
    float mean = stats[o * 2], var = stats[o * 2 + 1];
    float y = gam[o] * (v - mean) * rsqrtf(var + EPSV) + bet[o];
    y = fmaxf(y, 0.f);
    int b = pix / (h * w), ij = pix % (h * w);
    Y[(((size_t)b * 128 + o) * h + (ij / w)) * w + (ij % w)] = y;
}

// ---------------------------------------------------------------------------
// 7) final: out = relu(bn(y2) + bn(shortcut)), NCHW
// ---------------------------------------------------------------------------
__global__ void final_kernel(const float* __restrict__ Cy, const float* __restrict__ sy,
                             const float* __restrict__ g2, const float* __restrict__ b2,
                             const float* __restrict__ Cs, const float* __restrict__ ss,
                             const float* __restrict__ gs, const float* __restrict__ bs,
                             float* __restrict__ out, int B, int h, int w)
{
    int M = B * h * w;
    int idx = blockIdx.x * blockDim.x + threadIdx.x;
    if (idx >= M * 128) return;
    int pix = idx >> 7, o = idx & 127;
    float vy = Cy[idx];
    float vs = Cs[idx];
    float y = g2[o] * (vy - sy[o * 2]) * rsqrtf(sy[o * 2 + 1] + EPSV) + b2[o];
    float s = gs[o] * (vs - ss[o * 2]) * rsqrtf(ss[o * 2 + 1] + EPSV) + bs[o];
    float r = fmaxf(y + s, 0.f);
    int b = pix / (h * w), ij = pix % (h * w);
    out[(((size_t)b * 128 + o) * h + (ij / w)) * w + (ij % w)] = r;
}

// ---------------------------------------------------------------------------
// Host-side orchestration
// ---------------------------------------------------------------------------
extern "C" void kernel_launch(void* const* d_in, const int* in_sizes, int n_in,
                              void* d_out, int out_size, void* d_ws, size_t ws_size,
                              hipStream_t stream) {
    (void)in_sizes; (void)n_in; (void)out_size; (void)ws_size;

    const float* x    = (const float*)d_in[0];
    const float* wp1  = (const float*)d_in[1];
    const float* bp1  = (const float*)d_in[2];
    const float* wm1  = (const float*)d_in[3];
    const float* bm1  = (const float*)d_in[4];
    const float* wc1  = (const float*)d_in[5];
    const float* gam1 = (const float*)d_in[6];
    const float* bet1 = (const float*)d_in[7];
    const float* wp2  = (const float*)d_in[8];
    const float* bp2  = (const float*)d_in[9];
    const float* wm2  = (const float*)d_in[10];
    const float* bm2  = (const float*)d_in[11];
    const float* wc2  = (const float*)d_in[12];
    const float* gam2 = (const float*)d_in[13];
    const float* bet2 = (const float*)d_in[14];
    const float* wps  = (const float*)d_in[15];
    const float* bps  = (const float*)d_in[16];
    const float* wms  = (const float*)d_in[17];
    const float* bms  = (const float*)d_in[18];
    const float* wcs  = (const float*)d_in[19];
    const float* gams = (const float*)d_in[20];
    const float* bets = (const float*)d_in[21];

    // Problem constants
    const int B = 16, C1 = 64, H1 = 56, W1 = 56;
    const int h = 28, w = 28, C2 = 128;
    const int M = B * h * w;                 // 12544 = 784*16
    const int K1 = C1 * 9, nK32_1 = K1 / 32; // 576, 18
    const int K2 = C2 * 9, nK32_2 = K2 / 32; // 1152, 36
    const int Mt = M / 16;                   // 784

    // Workspace layout (bytes)
    char* ws = (char*)d_ws;
    unsigned short* Asw  = (unsigned short*)(ws + 0);            // 28,901,376 B
    unsigned short* Bsw  = (unsigned short*)(ws + 28901376);     //    294,912 B
    float* OM    = (float*)(ws + 29196288);                      //  1,354,752 B
    float* ConvS = (float*)(ws + 30551040);                      //  6,422,528 B
    float* ConvY = (float*)(ws + 36973568);                      //  6,422,528 B
    float* Y1    = (float*)(ws + 43396096);                      //  6,422,528 B
    float* st1   = (float*)(ws + 49818624);
    float* st2   = (float*)(ws + 49818624 + 1024);
    float* stS   = (float*)(ws + 49818624 + 2048);

    const int T = 256;
    dim3 gemmBlk(32, 8);

    // ---- shortcut DCN: x (stride 2) -> ConvS, stS ----
    offmask_kernel<<<(M * 27 + T - 1) / T, T, 0, stream>>>(x, wps, bps, wms, bms, OM,
                                                           B, C1, H1, W1, h, w, 2);
    gather_kernel<<<(M * 9 + T - 1) / T, T, 0, stream>>>(x, OM, Asw,
                                                         B, C1, H1, W1, h, w, 2, nK32_1);
    wprep_kernel<<<(K1 * 128 + T - 1) / T, T, 0, stream>>>(wcs, Bsw, K1, C1);
    gemm_wmma_kernel<<<Mt, gemmBlk, 0, stream>>>(Asw, Bsw, ConvS, nK32_1);
    stats_kernel<<<128, T, 0, stream>>>(ConvS, stS, M);

    // ---- DCN1: x (stride 2) -> ConvY -> bn+relu -> Y1 ----
    offmask_kernel<<<(M * 27 + T - 1) / T, T, 0, stream>>>(x, wp1, bp1, wm1, bm1, OM,
                                                           B, C1, H1, W1, h, w, 2);
    gather_kernel<<<(M * 9 + T - 1) / T, T, 0, stream>>>(x, OM, Asw,
                                                         B, C1, H1, W1, h, w, 2, nK32_1);
    wprep_kernel<<<(K1 * 128 + T - 1) / T, T, 0, stream>>>(wc1, Bsw, K1, C1);
    gemm_wmma_kernel<<<Mt, gemmBlk, 0, stream>>>(Asw, Bsw, ConvY, nK32_1);
    stats_kernel<<<128, T, 0, stream>>>(ConvY, st1, M);
    bnrelu_kernel<<<(M * 128 + T - 1) / T, T, 0, stream>>>(ConvY, st1, gam1, bet1, Y1,
                                                           B, h, w);

    // ---- DCN2: Y1 (stride 1) -> ConvY (reused) ----
    offmask_kernel<<<(M * 27 + T - 1) / T, T, 0, stream>>>(Y1, wp2, bp2, wm2, bm2, OM,
                                                           B, C2, h, w, h, w, 1);
    gather_kernel<<<(M * 9 + T - 1) / T, T, 0, stream>>>(Y1, OM, Asw,
                                                         B, C2, h, w, h, w, 1, nK32_2);
    wprep_kernel<<<(K2 * 128 + T - 1) / T, T, 0, stream>>>(wc2, Bsw, K2, C2);
    gemm_wmma_kernel<<<Mt, gemmBlk, 0, stream>>>(Asw, Bsw, ConvY, nK32_2);
    stats_kernel<<<128, T, 0, stream>>>(ConvY, st2, M);

    // ---- final: relu(bn(ConvY) + bn(ConvS)) -> d_out (NCHW f32) ----
    final_kernel<<<(M * 128 + T - 1) / T, T, 0, stream>>>(ConvY, st2, gam2, bet2,
                                                          ConvS, stS, gams, bets,
                                                          (float*)d_out, B, h, w);
}
